// Model_63453846831584
// MI455X (gfx1250) — matrix operand order repaired and checked by reading, not compiled, not run on hardware
//
#include <hip/hip_runtime.h>
#include <math.h>

#define B_ 256
#define S_ 512
#define D_ 768
#define H_ 384
#define N_ 768   // 2*H combined (Wo | Wr)
#define SEP_ID_ 103

typedef __attribute__((ext_vector_type(16))) __bf16 v16bf;
typedef __attribute__((ext_vector_type(8)))  float  v8f;

#if __has_builtin(__builtin_amdgcn_tanhf)
#define TANH(x) __builtin_amdgcn_tanhf(x)
#elif __has_builtin(__builtin_amdgcn_tanh_f32)
#define TANH(x) __builtin_amdgcn_tanh_f32(x)
#else
#define TANH(x) tanhf(x)
#endif

union Frag32 { uint4 q[2]; v16bf f; };

__device__ __forceinline__ v16bf mk16(uint4 lo, uint4 hi) {
  Frag32 f; f.q[0] = lo; f.q[1] = hi; return f.f;
}

// f32 -> bf16 bits, round-to-nearest-even
__device__ __forceinline__ unsigned short f2bf(float x) {
  unsigned int u = __float_as_uint(x);
  unsigned int r = (u + 0x7FFFu + ((u >> 16) & 1u)) >> 16;
  return (unsigned short)r;
}

// ---------------------------------------------------------------- find sep
__global__ void find_sep_kernel(const int* __restrict__ ids, int* __restrict__ sep) {
  __shared__ int smin;
  int b = blockIdx.x;
  if (threadIdx.x == 0) smin = S_;
  __syncthreads();
  for (int j = threadIdx.x; j < S_; j += blockDim.x)
    if (ids[b * S_ + j] == SEP_ID_) atomicMin(&smin, j);
  __syncthreads();
  if (threadIdx.x == 0) sep[b] = smin;
}

// ------------------------------------------- pack weights: Wct[n][k] bf16
__global__ void prep_kernel(const float* __restrict__ Wo, const float* __restrict__ bo,
                            const float* __restrict__ wow,
                            const float* __restrict__ Wr, const float* __restrict__ br,
                            const float* __restrict__ wrw,
                            unsigned short* __restrict__ Wct,
                            float* __restrict__ bc, float* __restrict__ wc) {
  int idx = blockIdx.x * blockDim.x + threadIdx.x;
  if (idx < N_ * D_) {
    int n = idx / D_, k = idx % D_;
    float v = (n < H_) ? Wo[k * H_ + n] : Wr[k * H_ + (n - H_)];
    Wct[(size_t)n * D_ + k] = f2bf(v);
  }
  if (idx < N_) {
    bc[idx] = (idx < H_) ? bo[idx] : br[idx - H_];
    wc[idx] = (idx < H_) ? wow[idx] : wrw[idx - H_];
  }
}

// ---------------------------------------------------------------- zeroing
__global__ void zero_kernel(float* __restrict__ scores, float* __restrict__ mixed) {
  int idx = blockIdx.x * blockDim.x + threadIdx.x;
  if (idx < 2 * B_ * S_) scores[idx] = 0.f;
  if (idx < B_ * 2 * D_) mixed[idx] = 0.f;
}

// ------------------------------------------------ main WMMA scores kernel
// grid: B * 4(Mtiles of 128 tokens) * 6(Ntiles of 128); block = 256 (8 waves)
__global__ __launch_bounds__(256) void scores_kernel(
    const float* __restrict__ lhs, const unsigned short* __restrict__ Wct,
    const float* __restrict__ bc, const float* __restrict__ wc,
    const int* __restrict__ sep, float* __restrict__ scores) {
  // double-buffered tiles; 80B row stride keeps b128 16B-aligned + conflict-free
  __shared__ unsigned short As[2][128][40];
  __shared__ unsigned short Bs[2][128][40];

  const int bid = blockIdx.x;
  const int nt  = bid % 6;
  const int mt  = (bid / 6) % 4;
  const int b   = bid / 24;
  const int tid = threadIdx.x;
  const int lane   = tid & 31;
  const int lane16 = lane & 15;
  const int hi     = lane >> 4;
  const int wid    = tid >> 5;
  const int waveM  = wid & 3;   // M offset = waveM*32
  const int waveN  = wid >> 2;  // N offset = waveN*64
  const int sepb = sep[b];

  // A staging geometry: 8 threads/row, 4 floats each, 4 row-blocks
  const int arow0 = tid >> 3;        // 0..31
  const int akq   = (tid & 7) * 4;   // 0..28
  const float* asrc[4];
  for (int rb = 0; rb < 4; ++rb) {
    int row = rb * 32 + arow0;
    int j = mt * 128 + row;
    int src = (j + 1 < sepb) ? (j + 1) : (j + 2);
    if (src > S_ - 1) src = S_ - 1;  // pad rows (masked later)
    asrc[rb] = lhs + ((size_t)(b * S_ + src)) * D_ + akq;
  }
  // B staging geometry: 2 threads/row, 16 bf16 (2 x b128) each
  const int brow  = tid >> 1;        // 0..127
  const int bhalf = (tid & 1) * 16;  // element offset 0 / 16
  const unsigned short* bsrc = Wct + (size_t)(nt * 128 + brow) * D_ + bhalf;

  v8f acc[2][4];
  for (int mi = 0; mi < 2; ++mi)
    for (int nf = 0; nf < 4; ++nf)
      for (int i = 0; i < 8; ++i) acc[mi][nf][i] = 0.f;

  auto stage = [&](int buf, int kk) {
    // A: gathered token rows, f32 -> bf16, via VGPRs
    for (int rb = 0; rb < 4; ++rb) {
      int row = rb * 32 + arow0;
      const float4 v = *(const float4*)(asrc[rb] + kk);
      union { unsigned short h[4]; uint2 u; } pk;
      pk.h[0] = f2bf(v.x); pk.h[1] = f2bf(v.y);
      pk.h[2] = f2bf(v.z); pk.h[3] = f2bf(v.w);
      *(uint2*)&As[buf][row][akq] = pk.u;
    }
    // B: already bf16 -> async copy straight into LDS (ASYNCcnt-tracked)
    unsigned int l0 = (unsigned int)(size_t)&Bs[buf][brow][bhalf];
    unsigned int l1 = l0 + 16;
    const unsigned short* g0 = bsrc + kk;
    const unsigned short* g1 = g0 + 8;
    asm volatile(
        "global_load_async_to_lds_b128 %0, %1, off\n\t"
        "global_load_async_to_lds_b128 %2, %3, off"
        :: "v"(l0), "v"(g0), "v"(l1), "v"(g1) : "memory");
  };

  stage(0, 0);
  for (int it = 0; it < D_ / 32; ++it) {
    const int cur = it & 1;
    asm volatile("s_wait_asynccnt 0x0" ::: "memory");
    __syncthreads();
    if (it + 1 < D_ / 32) stage(cur ^ 1, (it + 1) * 32);

    // fragment loads per ISA 16-bit A/B layouts
    v16bf afr[2], bfr[4];
    for (int mi = 0; mi < 2; ++mi) {
      const unsigned short* ar = &As[cur][waveM * 32 + mi * 16 + lane16][0];
      afr[mi] = mk16(*(const uint4*)(ar + hi * 8), *(const uint4*)(ar + hi * 8 + 16));
    }
    for (int nf = 0; nf < 4; ++nf) {
      const unsigned short* brp = &Bs[cur][waveN * 64 + nf * 16 + lane16][0];
      bfr[nf] = mk16(*(const uint4*)(brp + hi * 8), *(const uint4*)(brp + hi * 8 + 16));
    }
    for (int mi = 0; mi < 2; ++mi)
      for (int nf = 0; nf < 4; ++nf)
        acc[mi][nf] = __builtin_amdgcn_wmma_f32_16x16x32_bf16(
            false, afr[mi], false, bfr[nf], (short)0, acc[mi][nf], false, false);
  }

  // epilogue: score += sum_n tanh(acc + b[n]) * w[n]; N=384 boundary is tile-aligned
  const int poolBase = nt * 128 + waveN * 64;
  float* sc = scores + (size_t)((poolBase < H_) ? 0 : 1) * B_ * S_ + (size_t)b * S_;
  float bv[4], wv[4];
  for (int nf = 0; nf < 4; ++nf) {
    int ngl = poolBase + nf * 16 + lane16;
    bv[nf] = bc[ngl];
    wv[nf] = wc[ngl];
  }
  for (int mi = 0; mi < 2; ++mi) {
    float p[8];
    for (int i = 0; i < 8; ++i) p[i] = 0.f;
    for (int nf = 0; nf < 4; ++nf)
      for (int i = 0; i < 8; ++i)
        p[i] += TANH(acc[mi][nf][i] + bv[nf]) * wv[nf];
    for (int off = 1; off < 16; off <<= 1)
      for (int i = 0; i < 8; ++i) p[i] += __shfl_xor(p[i], off, 32);
    if (lane16 == 0) {  // lanes 0 (M rows +0..7) and 16 (M rows +8..15)
      int mbase = mt * 128 + waveM * 32 + mi * 16 + hi * 8;
      for (int i = 0; i < 8; ++i) atomicAdd(&sc[mbase + i], p[i]);
    }
  }
}

// --------------------------------------------------------------- softmax
__global__ void softmax_kernel(const float* __restrict__ scores,
                               const int* __restrict__ mask,
                               const int* __restrict__ sep,
                               float* __restrict__ att) {
  __shared__ float red[256];
  int b = blockIdx.x >> 1;
  int pool = blockIdx.x & 1;
  int tid = threadIdx.x;
  int sepb = sep[b];
  const float* sc = scores + (size_t)pool * B_ * S_ + (size_t)b * S_;
  float* at = att + (size_t)pool * B_ * S_ + (size_t)b * S_;

  float s[2]; int valid[2];
  for (int t = 0; t < 2; ++t) {
    int j = tid + t * 256;
    int ok = 0; float v = -INFINITY;
    if (j < S_ - 2) {
      int src = (j + 1 < sepb) ? (j + 1) : (j + 2);
      ok = mask[b * S_ + src];
      if (ok) v = sc[j];
    }
    valid[t] = ok; s[t] = v;
  }
  red[tid] = fmaxf(s[0], s[1]);
  __syncthreads();
  for (int sft = 128; sft > 0; sft >>= 1) {
    if (tid < sft) red[tid] = fmaxf(red[tid], red[tid + sft]);
    __syncthreads();
  }
  float mx = red[0];
  __syncthreads();
  float e[2];
  for (int t = 0; t < 2; ++t) e[t] = valid[t] ? __expf(s[t] - mx) : 0.f;
  red[tid] = e[0] + e[1];
  __syncthreads();
  for (int sft = 128; sft > 0; sft >>= 1) {
    if (tid < sft) red[tid] += red[tid + sft];
    __syncthreads();
  }
  float inv = 1.f / red[0];
  for (int t = 0; t < 2; ++t) at[tid + t * 256] = e[t] * inv;
}

// --------------------------- weighted pooling: both pools, one lhs pass
__global__ void pool_kernel(const float* __restrict__ lhs,
                            const float* __restrict__ att,
                            const int* __restrict__ sep,
                            float* __restrict__ mixed) {
  int b = blockIdx.x >> 2;
  int jc = blockIdx.x & 3;
  int tid = threadIdx.x;
  int sepb = sep[b];
  const float* ao = att + (size_t)b * S_;
  const float* ar = att + (size_t)B_ * S_ + (size_t)b * S_;
  float acco[3] = {0.f, 0.f, 0.f}, accr[3] = {0.f, 0.f, 0.f};
  for (int j = jc * 128; j < jc * 128 + 128; ++j) {
    float wo = ao[j], wr = ar[j];
    if (wo == 0.f && wr == 0.f) continue;
    int src = (j + 1 < sepb) ? (j + 1) : (j + 2);
    const float* rowp = lhs + ((size_t)(b * S_ + src)) * D_;
    for (int t = 0; t < 3; ++t) {
      float v = rowp[tid + t * 256];
      acco[t] += wo * v; accr[t] += wr * v;
    }
  }
  for (int t = 0; t < 3; ++t) {
    atomicAdd(&mixed[(size_t)b * (2 * D_) + tid + t * 256], acco[t]);
    atomicAdd(&mixed[(size_t)b * (2 * D_) + D_ + tid + t * 256], accr[t]);
  }
}

// ------------------------------------------------------------ final head
__global__ void head1_kernel(const float* __restrict__ mixed,
                             const float* __restrict__ W1,
                             const float* __restrict__ b1,
                             float* __restrict__ tbuf) {
  __shared__ float sm[2 * D_];
  int b = blockIdx.x / 3;
  int dc = blockIdx.x % 3;
  int tid = threadIdx.x;
  for (int i = tid; i < 2 * D_; i += 256) sm[i] = mixed[(size_t)b * 2 * D_ + i];
  __syncthreads();
  int d = dc * 256 + tid;
  float acc = b1[d];
#pragma unroll 4
  for (int k = 0; k < 2 * D_; ++k) acc += sm[k] * W1[(size_t)k * D_ + d];
  tbuf[(size_t)b * D_ + d] = acc;
}

__global__ void head2_kernel(const float* __restrict__ tbuf,
                             const float* __restrict__ Wf,
                             const float* __restrict__ bfv,
                             float* __restrict__ out) {
  __shared__ float red[256];
  int b = blockIdx.x;
  int tid = threadIdx.x;
  float pc[6] = {0.f, 0.f, 0.f, 0.f, 0.f, 0.f};
  for (int d = tid; d < D_; d += 256) {
    float tv = tbuf[(size_t)b * D_ + d];
    for (int c = 0; c < 6; ++c) pc[c] += tv * Wf[d * 6 + c];
  }
  for (int c = 0; c < 6; ++c) {
    red[tid] = pc[c];
    __syncthreads();
    for (int s = 128; s > 0; s >>= 1) {
      if (tid < s) red[tid] += red[tid + s];
      __syncthreads();
    }
    if (tid == 0) out[b * 6 + c] = red[0] + bfv[c];
    __syncthreads();
  }
}

// ---------------------------------------------------------------- launch
extern "C" void kernel_launch(void* const* d_in, const int* in_sizes, int n_in,
                              void* d_out, int out_size, void* d_ws, size_t ws_size,
                              hipStream_t stream) {
  const float* lhs = (const float*)d_in[0];
  const int* ids   = (const int*)d_in[1];
  const int* mask  = (const int*)d_in[2];
  const float* Wo  = (const float*)d_in[3];
  const float* bo  = (const float*)d_in[4];
  const float* wow = (const float*)d_in[5];
  const float* Wr  = (const float*)d_in[6];
  const float* br  = (const float*)d_in[7];
  const float* wrw = (const float*)d_in[8];
  const float* W1  = (const float*)d_in[9];
  const float* b1  = (const float*)d_in[10];
  const float* Wf  = (const float*)d_in[11];
  const float* bfv = (const float*)d_in[12];
  float* out = (float*)d_out;

  char* p = (char*)d_ws;
  auto take = [&](size_t bytes) { char* r = p; p += (bytes + 255) & ~size_t(255); return r; };
  unsigned short* Wct = (unsigned short*)take((size_t)N_ * D_ * 2);
  float* bc     = (float*)take((size_t)N_ * 4);
  float* wc     = (float*)take((size_t)N_ * 4);
  int*   sep    = (int*)take((size_t)B_ * 4);
  float* scores = (float*)take((size_t)2 * B_ * S_ * 4);
  float* att    = (float*)take((size_t)2 * B_ * S_ * 4);
  float* mixed  = (float*)take((size_t)B_ * 2 * D_ * 4);
  float* tbuf   = (float*)take((size_t)B_ * D_ * 4);

  find_sep_kernel<<<B_, 256, 0, stream>>>(ids, sep);
  prep_kernel<<<(N_ * D_ + 255) / 256, 256, 0, stream>>>(Wo, bo, wow, Wr, br, wrw, Wct, bc, wc);
  zero_kernel<<<(B_ * 2 * D_ + 255) / 256, 256, 0, stream>>>(scores, mixed);
  scores_kernel<<<B_ * 4 * 6, 256, 0, stream>>>(lhs, Wct, bc, wc, sep, scores);
  softmax_kernel<<<B_ * 2, 256, 0, stream>>>(scores, mask, sep, att);
  pool_kernel<<<B_ * 4, 256, 0, stream>>>(lhs, att, sep, mixed);
  head1_kernel<<<B_ * 3, 256, 0, stream>>>(mixed, W1, b1, tbuf);
  head2_kernel<<<B_, 256, 0, stream>>>(tbuf, Wf, bfv, out);
}
